// ModulatedConv2d_910533066889
// MI455X (gfx1250) — matrix operand order repaired and checked by reading, not compiled, not run on hardware
//
#include <hip/hip_runtime.h>

typedef __attribute__((ext_vector_type(16))) _Float16 v16h;
typedef __attribute__((ext_vector_type(8)))  float    v8f;

#define BN    8
#define CIN   32
#define COUT  32
#define HDIM  512
#define WDIM  512
#define SDIM  512
#define KSTEPS 9          // 3x3 spatial taps; K-order = tap*32 + cin
#define TH    4           // tile rows
#define TW    32          // tile cols  (TH*TW = 128 pixels = 8 waves * 16)
#define LROW  (TW + 2)    // 34
#define AFRAG_PER_B (2 * KSTEPS * 32 * 16)   // halfs per batch: mtile*kstep*lane*e

// ---------------------------------------------------------------------------
// Prep: s = style @ (mod_w/sqrt(S))^T + mod_b ; wmod = cs*w*s ; demod ;
// then scatter modulated weights into WMMA A-fragment layout (f16) in ws.
// wsA[b][mtile][kstep][lane][e], 16 halfs per lane = one v16h A operand.
// A order used here (contiguous k): lane<16 -> row M=lane, K = {0..15};
//                        lane>=16 -> row M=lane-16, K = {16..31}.
// ---------------------------------------------------------------------------
__global__ __launch_bounds__(256) void modconv_prep(
    const float* __restrict__ style, const float* __restrict__ weight,
    const float* __restrict__ mod_w, const float* __restrict__ mod_b,
    _Float16* __restrict__ wsA)
{
    const int b   = blockIdx.x;
    const int tid = threadIdx.x;
    __shared__ float s_mod[CIN];
    __shared__ float s_dem[COUT];

    const float mod_scale  = 0.044194173824159216f;  // 1/sqrt(512)
    const float conv_scale = 0.05892556509887896f;   // 1/sqrt(32*9)

    if (tid < CIN) {
        float acc = 0.f;
        const float* st = style + (size_t)b * SDIM;
        const float* mw = mod_w + (size_t)tid * SDIM;
        for (int j = 0; j < SDIM; ++j) acc += st[j] * mw[j];
        s_mod[tid] = acc * mod_scale + mod_b[tid];
    }
    __syncthreads();

    if (tid < COUT) {
        float ss = 0.f;
        const float* wco = weight + (size_t)tid * CIN * 9;
        for (int c = 0; c < CIN; ++c) {
            float sm = s_mod[c] * conv_scale;
            for (int t = 0; t < 9; ++t) {
                float v = wco[c * 9 + t] * sm;
                ss += v * v;
            }
        }
        s_dem[tid] = rsqrtf(ss + 1e-8f);
    }
    __syncthreads();

    for (int idx = tid; idx < AFRAG_PER_B; idx += 256) {
        int e    = idx & 15;
        int lane = (idx >> 4) & 31;
        int ms   = idx >> 9;          // mtile*9 + kstep
        int s    = ms % KSTEPS;
        int mt   = ms / KSTEPS;
        int mrow = lane & 15;
        int off  = (lane >= 16) ? 16 : 0;
        int cin  = off + e;   // K within step == cin
        int cout = mt * 16 + mrow;
        int ky = s / 3, kx = s % 3;
        float v = conv_scale * weight[((size_t)cout * CIN + cin) * 9 + ky * 3 + kx]
                * s_mod[cin] * s_dem[cout];
        wsA[(size_t)b * AFRAG_PER_B + idx] = (_Float16)v;
    }
}

// ---------------------------------------------------------------------------
// Conv: implicit GEMM, M=cout(2x16), N=pixels, K=288 in 9 steps of 32 (f16).
// LDS x-tile layout [yi][xi][cin] so each B lane reads 32 contiguous bytes:
// B order used here (contiguous k): lane<16 -> col N=lane, K=0..15 (cin 0..15);
//                        lane>=16 -> col N=lane-16, K=16..31 (cin 16..31).
// ---------------------------------------------------------------------------
__global__ __launch_bounds__(256) void modconv_wmma(
    const float* __restrict__ x, const _Float16* __restrict__ wsA,
    float* __restrict__ out)
{
    const int b  = blockIdx.z;
    const int y0 = blockIdx.y * TH;
    const int x0 = blockIdx.x * TW;
    const int tid = threadIdx.x;

    __shared__ _Float16 lx[(TH + 2) * LROW * CIN];   // 13,056 B

    // --- stage halo tile (f32 -> f16) ---
    const float* xb = x + (size_t)b * CIN * HDIM * WDIM;
    for (int idx = tid; idx < CIN * (TH + 2) * LROW; idx += 256) {
        int xi  = idx % LROW;
        int rem = idx / LROW;
        int yi  = rem % (TH + 2);
        int c   = rem / (TH + 2);
        int gy = y0 - 1 + yi;
        int gx = x0 - 1 + xi;
        float v = 0.f;
        if (gy >= 0 && gy < HDIM && gx >= 0 && gx < WDIM)
            v = xb[(size_t)c * HDIM * WDIM + (size_t)gy * WDIM + gx];
        lx[(yi * LROW + xi) * CIN + c] = (_Float16)v;
    }
    __syncthreads();

    const int wave = tid >> 5;
    const int lane = tid & 31;
    const int r    = wave >> 1;            // row in tile 0..3
    const int xh   = (wave & 1) * 16;      // col half 0/16
    const int p    = lane & 15;            // pixel (N) index
    const int coff = (lane >= 16) ? 16 : 0;// B cin base per lane half

    v8f acc0 = {};
    v8f acc1 = {};

    const _Float16* Abase = wsA + (size_t)b * AFRAG_PER_B + (size_t)lane * 16;

#pragma unroll
    for (int s = 0; s < KSTEPS; ++s) {
        const int ky = s / 3, kx = s % 3;
        const int yi = r + ky;
        const int xi = xh + p + kx;
        v16h bfrag = *(const v16h*)&lx[(yi * LROW + xi) * CIN + coff];
        v16h a0 = *(const v16h*)(Abase + (size_t)(0 * KSTEPS + s) * 512);
        v16h a1 = *(const v16h*)(Abase + (size_t)(1 * KSTEPS + s) * 512);
        acc0 = __builtin_amdgcn_wmma_f32_16x16x32_f16(
            false, a0, false, bfrag, (short)0, acc0, false, false);
        acc1 = __builtin_amdgcn_wmma_f32_16x16x32_f16(
            false, a1, false, bfrag, (short)0, acc1, false, false);
    }

    // --- store D (lane -> N, vgpr r -> M, lane half -> M+8) ---
    const int px    = x0 + xh + p;
    const int py    = y0 + r;
    const int mbase = (lane >= 16) ? 8 : 0;
    float* ob = out + (size_t)b * COUT * HDIM * WDIM + (size_t)py * WDIM + px;
#pragma unroll
    for (int m = 0; m < 8; ++m) {
        ob[(size_t)(mbase + m) * HDIM * WDIM]      = acc0[m];
        ob[(size_t)(16 + mbase + m) * HDIM * WDIM] = acc1[m];
    }
}

extern "C" void kernel_launch(void* const* d_in, const int* in_sizes, int n_in,
                              void* d_out, int out_size, void* d_ws, size_t ws_size,
                              hipStream_t stream) {
    const float* x      = (const float*)d_in[0];
    const float* style  = (const float*)d_in[1];
    const float* weight = (const float*)d_in[2];
    const float* mod_w  = (const float*)d_in[3];
    const float* mod_b  = (const float*)d_in[4];
    float* out = (float*)d_out;
    _Float16* wsA = (_Float16*)d_ws;   // needs BN*AFRAG_PER_B*2 = 147,456 B

    modconv_prep<<<dim3(BN), 256, 0, stream>>>(style, weight, mod_w, mod_b, wsA);
    modconv_wmma<<<dim3(WDIM / TW, HDIM / TH, BN), 256, 0, stream>>>(x, wsA, out);
}
